// Speller_35201551958642
// MI455X (gfx1250) — matrix operand order repaired and checked by reading, not compiled, not run on hardware
//
#include <hip/hip_runtime.h>

// ---------------- problem constants ----------------
#define HH   256
#define QQ   64
#define DD   512
#define VV   34
#define LL   1024
#define NN   64
#define TT   256
#define MENC (LL * NN)        // 65536 encoder rows
#define HSZ  (NN * HH)        // 64*256 per-layer state

typedef __bf16 bf16_t;
typedef __attribute__((ext_vector_type(16))) __bf16 v16bf;
typedef __attribute__((ext_vector_type(8)))  __bf16 v8bf;
typedef __attribute__((ext_vector_type(8)))  float  v8f;

// ---------------- WMMA fragment loaders (ISA 7.12.2 layouts, wave32) ----------------
// A (16x32 bf16): lane l holds row M=l%16; half=l/16; slots 0..7 -> K = kb+half*8+j,
// slots 8..15 -> K = kb+16+half*8+j.
__device__ __forceinline__ v16bf load_afrag_bf(const bf16_t* base, int stride,
                                               int mrow, int kb, int half) {
  const bf16_t* p = base + (size_t)mrow * stride + kb + half * 8;
  v8bf lo = *(const v8bf*)p;
  v8bf hi = *(const v8bf*)(p + 16);
  v16bf a;
#pragma unroll
  for (int i = 0; i < 8; ++i) { a[i] = lo[i]; a[8 + i] = hi[i]; }
  return a;
}

__device__ __forceinline__ v16bf load_afrag_f32(const float* base, int stride,
                                                int mrow, int kb, int half) {
  const float* p = base + (size_t)mrow * stride + kb + half * 8;
  v8f lo = *(const v8f*)p;
  v8f hi = *(const v8f*)(p + 16);
  v16bf a;
#pragma unroll
  for (int i = 0; i < 8; ++i) { a[i] = (bf16_t)lo[i]; a[8 + i] = (bf16_t)hi[i]; }
  return a;
}

// B (32x16 bf16): lane l holds col N=l%16; half=l/16; slots 0..7 -> K = kb+half*8+j,
// slots 8..15 -> K = kb+16+half*8+j. W is row-major [Ntot][K].
__device__ __forceinline__ v16bf load_bfrag(const bf16_t* W, int ldw,
                                            int ncol, int kb, int half) {
  const bf16_t* p = W + (size_t)ncol * ldw + kb + half * 8;
  v8bf lo = *(const v8bf*)p;
  v8bf hi = *(const v8bf*)(p + 16);
  v16bf b;
#pragma unroll
  for (int i = 0; i < 8; ++i) { b[i] = lo[i]; b[8 + i] = hi[i]; }
  return b;
}

__device__ __forceinline__ float sigm(float x) { return 1.f / (1.f + __expf(-x)); }
__device__ __forceinline__ float eluf(float x) { return x > 0.f ? x : (__expf(x) - 1.f); }

// ---------------- software barrier across the 8 recurrence WGs ----------------
__device__ void sw_barrier(unsigned* cnt, unsigned* gen, unsigned nwg) {
  __threadfence();
  __syncthreads();
  if (threadIdx.x == 0) {
    unsigned g = __hip_atomic_load(gen, __ATOMIC_ACQUIRE, __HIP_MEMORY_SCOPE_AGENT);
    unsigned prev = __hip_atomic_fetch_add(cnt, 1u, __ATOMIC_ACQ_REL, __HIP_MEMORY_SCOPE_AGENT);
    if (prev == nwg - 1) {
      __hip_atomic_store(cnt, 0u, __ATOMIC_RELAXED, __HIP_MEMORY_SCOPE_AGENT);
      __hip_atomic_fetch_add(gen, 1u, __ATOMIC_RELEASE, __HIP_MEMORY_SCOPE_AGENT);
    } else {
      while (__hip_atomic_load(gen, __ATOMIC_ACQUIRE, __HIP_MEMORY_SCOPE_AGENT) == g) {
        __builtin_amdgcn_s_sleep(2);
      }
    }
  }
  __syncthreads();
  __threadfence();
}

// ---------------- prep kernels ----------------
__global__ void prep_wcat(const float* Wih, const float* Whh, int Kin, bf16_t* dst, int total) {
  int K = Kin + HH;
  for (int i = blockIdx.x * blockDim.x + threadIdx.x; i < total; i += gridDim.x * blockDim.x) {
    int nrow = i / K, k = i % K;
    float v = (k < Kin) ? Wih[(size_t)nrow * Kin + k] : Whh[(size_t)nrow * HH + (k - Kin)];
    dst[i] = (bf16_t)v;
  }
}

__global__ void prep_cvt(const float* src, bf16_t* dst, int total) {
  for (int i = blockIdx.x * blockDim.x + threadIdx.x; i < total; i += gridDim.x * blockDim.x)
    dst[i] = (bf16_t)src[i];
}

__global__ void prep_bias(const float* bih, const float* bhh, float* dst) {
  int i = blockIdx.x * blockDim.x + threadIdx.x;
  if (i < 4 * HH) dst[i] = bih[i] + bhh[i];
}

__global__ void prep_char(const int* labels, const float* emb, bf16_t* dst, int total) {
  for (int i = blockIdx.x * blockDim.x + threadIdx.x; i < total; i += gridDim.x * blockDim.x) {
    int tn = i / HH, h = i % HH;
    dst[i] = (bf16_t)emb[(size_t)labels[tn] * HH + h];
  }
}

__global__ void prep_state(const float* inith, const float* initc, bf16_t* h0, float* c0, unsigned* bar) {
  int i = blockIdx.x * blockDim.x + threadIdx.x;
  if (i < 3 * HSZ) {
    int layer = i / HSZ, h = i % HH;
    h0[i] = (bf16_t)inith[layer * HH + h];
    c0[i] = initc[layer * HH + h];
  }
  if (i < 8) bar[i] = 0u;
}

// ---------------- encoder GEMM: C[M,Ntot] = act(A[M,K] @ W[Ntot,K]^T + bias) ----------------
// Each wave computes a 16x64 strip (4 N-tiles reuse one A fragment).
// mode 0: store bf16 row-major [M][Ntot]  (hidden)
// mode 1: store f32 key layout   [nb][L][64]  (m = l*64+nb, n = q)
// mode 2: store bf16 value layout [nb][L][256] (m = l*64+nb, n = h)
__global__ __launch_bounds__(256) void enc_gemm(
    const float* Af, const bf16_t* Ab, int lda,
    const bf16_t* W, int ldw, const float* bias,
    bf16_t* outB, float* outF,
    int Ntot, int K, int mode, int doElu) {
  const int lane = threadIdx.x & 31;
  const int wave = (blockIdx.x * blockDim.x + threadIdx.x) >> 5;
  const int nstrips = Ntot >> 6;
  const int mt = wave / nstrips;
  const int ns = wave % nstrips;
  const int half = lane >> 4;
  const int mrow = (mt << 4) + (lane & 15);

  v8f acc[4] = {};
  for (int kb = 0; kb < K; kb += 32) {
    v16bf a = Af ? load_afrag_f32(Af, lda, mrow, kb, half)
                 : load_afrag_bf(Ab, lda, mrow, kb, half);
    if (kb + 32 < K) {
      if (Af) __builtin_prefetch(Af + (size_t)mrow * lda + kb + 32, 0, 1);
      else    __builtin_prefetch(Ab + (size_t)mrow * lda + kb + 32, 0, 1);
    }
#pragma unroll
    for (int t = 0; t < 4; ++t) {
      int ncol = (ns << 6) + (t << 4) + (lane & 15);
      v16bf b = load_bfrag(W, ldw, ncol, kb, half);
      acc[t] = __builtin_amdgcn_wmma_f32_16x16x32_bf16(false, a, false, b,
                                                       (short)0, acc[t], false, false);
    }
  }
#pragma unroll
  for (int t = 0; t < 4; ++t) {
    int n = (ns << 6) + (t << 4) + (lane & 15);
    float bv = bias ? bias[n] : 0.f;
#pragma unroll
    for (int r = 0; r < 8; ++r) {
      int mm = (mt << 4) + r + (half << 3);
      float v = acc[t][r] + bv;
      if (doElu) v = eluf(v);
      if (mode == 0) {
        outB[(size_t)mm * Ntot + n] = (bf16_t)v;
      } else {
        int l = mm >> 6, nb = mm & 63;
        if (mode == 1) outF[((size_t)nb * LL + l) * QQ + n] = v;
        else           outB[((size_t)nb * LL + l) * HH + n] = (bf16_t)v;
      }
    }
  }
}

// ---------------- decoder recurrence: 3 LSTM cells, one step ----------------
// 8 WGs x 8 waves = 64 waves; wave (mt, ht) owns rows mt*16..+15, hidden cols ht*16..+15,
// and computes all four gate tiles (i,f,g,o) so the nonlinearity runs out of registers.
__global__ __launch_bounds__(256) void dec_recurrence(
    const bf16_t* wcat0, const bf16_t* wcat1, const bf16_t* wcat2,
    const float* bias3,          // [3][1024]
    const bf16_t* ctx,           // [64][256]
    const bf16_t* chart,         // [64][256] (this step's char embedding)
    const bf16_t* hprev,         // [3][64][256]  (t-1)
    bf16_t* hnew,                // [3][64][256]  (t)
    float* cbuf,                 // [3][64][256]
    unsigned* bcnt, unsigned* bgen) {
  const int lane = threadIdx.x & 31;
  const int gid  = blockIdx.x * 8 + (threadIdx.x >> 5);   // 0..63
  const int mt   = gid >> 4;
  const int ht   = gid & 15;
  const int half = lane >> 4;
  const int mrow = (mt << 4) + (lane & 15);
  const int hc   = (ht << 4) + (lane & 15);

  for (int cell = 0; cell < 3; ++cell) {
    const bf16_t* W = (cell == 0) ? wcat0 : (cell == 1) ? wcat1 : wcat2;
    const int Kc = (cell == 0) ? 768 : 512;
    v8f acc[4] = {};
    for (int kb = 0; kb < Kc; kb += 32) {
      const bf16_t* src; int ko;
      if (cell == 0) {
        if (kb < 256)      { src = ctx;            ko = kb; }
        else if (kb < 512) { src = chart;          ko = kb - 256; }
        else               { src = hprev;          ko = kb - 512; }        // h0(t-1)
      } else if (cell == 1) {
        if (kb < 256)      { src = hnew;           ko = kb; }              // h0(t)
        else               { src = hprev + HSZ;    ko = kb - 256; }        // h1(t-1)
      } else {
        if (kb < 256)      { src = hnew + HSZ;     ko = kb; }              // h1(t)
        else               { src = hprev + 2*HSZ;  ko = kb - 256; }        // h2(t-1)
      }
      v16bf a = load_afrag_bf(src, HH, mrow, ko, half);
#pragma unroll
      for (int g = 0; g < 4; ++g) {
        int ncol = (g << 8) + hc;                 // gate g, hidden col hc
        v16bf b = load_bfrag(W, Kc, ncol, kb, half);
        acc[g] = __builtin_amdgcn_wmma_f32_16x16x32_bf16(false, a, false, b,
                                                         (short)0, acc[g], false, false);
      }
    }
    const float bi = bias3[cell * 1024 + 0   + hc];
    const float bfv = bias3[cell * 1024 + 256 + hc];
    const float bg = bias3[cell * 1024 + 512 + hc];
    const float bo = bias3[cell * 1024 + 768 + hc];
#pragma unroll
    for (int r = 0; r < 8; ++r) {
      int mm = (mt << 4) + r + (half << 3);
      float ig = sigm(acc[0][r] + bi);
      float fg = sigm(acc[1][r] + bfv);
      float gg = tanhf(acc[2][r] + bg);
      float og = sigm(acc[3][r] + bo);
      size_t idx = (size_t)cell * HSZ + (size_t)mm * HH + hc;
      float cn = fg * cbuf[idx] + ig * gg;
      cbuf[idx] = cn;
      hnew[idx] = (bf16_t)(og * tanhf(cn));
    }
    if (cell < 2) sw_barrier(bcnt, bgen, 8);
  }
}

// ---------------- attention (+ q-MLP + output projection), one WG per batch row ----------------
__global__ __launch_bounds__(256) void dec_attention(
    const bf16_t* h2,        // [64][256]  (layer-2 h of this step)
    const float* qW0, const float* qb0, const float* qW1, const float* qb1,
    const float* key,        // [64][1024][64] f32
    const bf16_t* value,     // [64][1024][256] bf16
    const int* seq_lens,
    const float* outW, const float* outb,
    bf16_t* ctx_out,         // [64][256]
    float* out)              // [64][34] for this step, or nullptr
{
  __shared__ float h2f[HH], hid[HH], qv[QQ], wls[LL], red[256];
  const int n = blockIdx.x, tid = threadIdx.x;

  h2f[tid] = (float)h2[n * HH + tid];
  __syncthreads();

  { // hidden = elu(h2 @ qW0^T + qb0)
    float s = qb0[tid];
    const float* wr = qW0 + (size_t)tid * HH;
    for (int k = 0; k < HH; ++k) s += h2f[k] * wr[k];
    hid[tid] = eluf(s);
  }
  __syncthreads();
  if (tid < QQ) { // q = hidden @ qW1^T + qb1
    float s = qb1[tid];
    const float* wr = qW1 + (size_t)tid * HH;
    for (int k = 0; k < HH; ++k) s += hid[k] * wr[k];
    qv[tid] = s;
  }
  __syncthreads();

  const int len = seq_lens[n];
  float sloc[4];
  float lmax = -3.4e38f;
#pragma unroll
  for (int i = 0; i < 4; ++i) {
    int l = i * 256 + tid;
    const float* kr = key + ((size_t)n * LL + l) * QQ;
    float s = 0.f;
    for (int q = 0; q < QQ; ++q) s += kr[q] * qv[q];
    sloc[i] = s;
    lmax = fmaxf(lmax, s);
  }
  red[tid] = lmax; __syncthreads();
  for (int s = 128; s > 0; s >>= 1) { if (tid < s) red[tid] = fmaxf(red[tid], red[tid + s]); __syncthreads(); }
  const float gmax = red[0]; __syncthreads();

  float psum = 0.f;
#pragma unroll
  for (int i = 0; i < 4; ++i) {
    int l = i * 256 + tid;
    float p = (l < len) ? __expf(sloc[i] - gmax) : 0.f;
    wls[l] = p;
    psum += p;
  }
  red[tid] = psum; __syncthreads();
  for (int s = 128; s > 0; s >>= 1) { if (tid < s) red[tid] += red[tid + s]; __syncthreads(); }
  const float inv = 1.f / red[0];

  { // ctx[h] = sum_l w_l * value[n][l][h] * inv
    float acc = 0.f;
    const bf16_t* vb = value + (size_t)n * LL * HH + tid;
    for (int l = 0; l < LL; ++l) acc += wls[l] * (float)vb[(size_t)l * HH];
    float cx = acc * inv;
    ctx_out[n * HH + tid] = (bf16_t)cx;
    hid[tid] = cx;   // reuse as ctx f32 for the output projection
  }
  __syncthreads();

  if (out && tid < VV) {
    float s = outb[tid];
    const float* wr = outW + (size_t)tid * (2 * HH);
    for (int k = 0; k < HH; ++k) s += hid[k] * wr[k];
    for (int k = 0; k < HH; ++k) s += h2f[k] * wr[HH + k];
    out[n * VV + tid] = s;
  }
}

// ---------------- host orchestration ----------------
extern "C" void kernel_launch(void* const* d_in, const int* in_sizes, int n_in,
                              void* d_out, int out_size, void* d_ws, size_t ws_size,
                              hipStream_t stream) {
  const float* seqs  = (const float*)d_in[0];
  const int*   slen  = (const int*)  d_in[1];
  const int*   labs  = (const int*)  d_in[2];
  const float* emb   = (const float*)d_in[3];
  const float* inith = (const float*)d_in[4];
  const float* initc = (const float*)d_in[5];
  const float *Wih[3], *Whh[3], *bih[3], *bhh[3];
  for (int i = 0; i < 3; ++i) {
    Wih[i] = (const float*)d_in[6 + 4*i]; Whh[i] = (const float*)d_in[7 + 4*i];
    bih[i] = (const float*)d_in[8 + 4*i]; bhh[i] = (const float*)d_in[9 + 4*i];
  }
  const float* qW0 = (const float*)d_in[18]; const float* qb0 = (const float*)d_in[19];
  const float* qW1 = (const float*)d_in[20]; const float* qb1 = (const float*)d_in[21];
  const float* kW0 = (const float*)d_in[22]; const float* kb0 = (const float*)d_in[23];
  const float* kW1 = (const float*)d_in[24]; const float* kb1 = (const float*)d_in[25];
  const float* vW0 = (const float*)d_in[26]; const float* vb0 = (const float*)d_in[27];
  const float* vW1 = (const float*)d_in[28]; const float* vb1 = (const float*)d_in[29];
  const float* outW = (const float*)d_in[30]; const float* outb = (const float*)d_in[31];
  float* out = (float*)d_out;

  // ---- workspace carve-up ----
  char* base = (char*)d_ws;
  size_t off = 0;
  auto take = [&](size_t bytes) { off = (off + 255) & ~(size_t)255; size_t r = off; off += bytes; return r; };
  bf16_t* wcat0 = (bf16_t*)(base + take((size_t)1024 * 768 * 2));
  bf16_t* wcat1 = (bf16_t*)(base + take((size_t)1024 * 512 * 2));
  bf16_t* wcat2 = (bf16_t*)(base + take((size_t)1024 * 512 * 2));
  float*  bias3 = (float*) (base + take((size_t)3 * 1024 * 4));
  bf16_t* kw0b  = (bf16_t*)(base + take((size_t)DD * DD * 2));
  bf16_t* kw1b  = (bf16_t*)(base + take((size_t)QQ * DD * 2));
  bf16_t* vw0b  = (bf16_t*)(base + take((size_t)HH * DD * 2));
  bf16_t* vw1b  = (bf16_t*)(base + take((size_t)HH * HH * 2));
  bf16_t* charb = (bf16_t*)(base + take((size_t)TT * NN * HH * 2));
  bf16_t* hidden= (bf16_t*)(base + take((size_t)MENC * DD * 2));
  float*  keyf  = (float*) (base + take((size_t)NN * LL * QQ * 4));
  bf16_t* valb  = (bf16_t*)(base + take((size_t)NN * LL * HH * 2));
  bf16_t* ctxb  = (bf16_t*)(base + take((size_t)NN * HH * 2));
  bf16_t* hbuf  = (bf16_t*)(base + take((size_t)2 * 3 * HSZ * 2));
  float*  cbuf  = (float*) (base + take((size_t)3 * HSZ * 4));
  unsigned* bar = (unsigned*)(base + take(256));
  unsigned* bcnt = bar, *bgen = bar + 1;

  // ---- prep ----
  prep_wcat<<<1024, 256, 0, stream>>>(Wih[0], Whh[0], 512, wcat0, 1024 * 768);
  prep_wcat<<<1024, 256, 0, stream>>>(Wih[1], Whh[1], 256, wcat1, 1024 * 512);
  prep_wcat<<<1024, 256, 0, stream>>>(Wih[2], Whh[2], 256, wcat2, 1024 * 512);
  prep_bias<<<4, 256, 0, stream>>>(bih[0], bhh[0], bias3);
  prep_bias<<<4, 256, 0, stream>>>(bih[1], bhh[1], bias3 + 1024);
  prep_bias<<<4, 256, 0, stream>>>(bih[2], bhh[2], bias3 + 2048);
  prep_cvt<<<512, 256, 0, stream>>>(kW0, kw0b, DD * DD);
  prep_cvt<<<128, 256, 0, stream>>>(kW1, kw1b, QQ * DD);
  prep_cvt<<<512, 256, 0, stream>>>(vW0, vw0b, HH * DD);
  prep_cvt<<<256, 256, 0, stream>>>(vW1, vw1b, HH * HH);
  prep_char<<<2048, 256, 0, stream>>>(labs, emb, charb, TT * NN * HH);
  prep_state<<<192, 256, 0, stream>>>(inith, initc, hbuf, cbuf, bar);

  // ---- encoder: key = MLP(seqs), value = MLP(seqs) ----
  // GEMM grids: waves = (M/16)*(N/64), 8 waves per WG
  enc_gemm<<<(MENC / 16) * (512 / 64) / 8, 256, 0, stream>>>(
      seqs, nullptr, DD, kw0b, DD, kb0, hidden, nullptr, 512, DD, 0, 1);
  enc_gemm<<<(MENC / 16) * (64 / 64) / 8, 256, 0, stream>>>(
      nullptr, hidden, DD, kw1b, DD, kb1, nullptr, keyf, 64, DD, 1, 0);
  enc_gemm<<<(MENC / 16) * (256 / 64) / 8, 256, 0, stream>>>(
      seqs, nullptr, DD, vw0b, DD, vb0, hidden, nullptr, 256, DD, 0, 1);
  enc_gemm<<<(MENC / 16) * (256 / 64) / 8, 256, 0, stream>>>(
      nullptr, hidden, HH, vw1b, HH, vb1, valb, nullptr, 256, HH, 2, 0);

  // ---- initial context from initial h2 (no output row) ----
  dec_attention<<<NN, 256, 0, stream>>>(hbuf + 2 * HSZ, qW0, qb0, qW1, qb1,
                                        keyf, valb, slen, outW, outb, ctxb, nullptr);

  // ---- decode loop: T steps, 2 launches per step ----
  for (int t = 0; t < TT; ++t) {
    int p = t & 1;
    const bf16_t* hp = hbuf + (size_t)p * 3 * HSZ;
    bf16_t*       hn = hbuf + (size_t)(1 - p) * 3 * HSZ;
    dec_recurrence<<<8, 256, 0, stream>>>(wcat0, wcat1, wcat2, bias3,
                                          ctxb, charb + (size_t)t * NN * HH,
                                          hp, hn, cbuf, bcnt, bgen);
    dec_attention<<<NN, 256, 0, stream>>>(hn + 2 * HSZ, qW0, qb0, qW1, qb1,
                                          keyf, valb, slen, outW, outb,
                                          ctxb, out + (size_t)t * NN * VV);
  }
}
